// PointNetPPEncoderFP_58299886076321
// MI455X (gfx1250) — matrix operand order repaired and checked by reading, not compiled, not run on hardware
//
#include <hip/hip_runtime.h>
#include <hip/hip_bf16.h>
#include <math.h>

typedef __attribute__((ext_vector_type(16))) _Float16 v16h;
typedef __attribute__((ext_vector_type(8)))  _Float16 v8h;
typedef __attribute__((ext_vector_type(8)))  float    v8f;
typedef __attribute__((ext_vector_type(4)))  unsigned int v4u;
typedef __attribute__((ext_vector_type(8)))  int      v8i;
typedef __attribute__((ext_vector_type(4)))  int      v4i;

#if __has_builtin(__builtin_amdgcn_tensor_load_to_lds) && __has_builtin(__builtin_amdgcn_s_wait_tensorcnt)
#define USE_TDM 1
#else
#define USE_TDM 0
#endif
#if __has_include(<hip/amd_detail/amd_gfx1250_TDM.h>)
#define TDM_SIX_ARG 1
#else
#define TDM_SIX_ARG 0
#endif

#define BB   16
#define NN   8192
#define CEMB 128
#define SA1N 512
#define SA2N 128
#define LT0  (BB*NN)    // 131072
#define LT1  (BB*SA1N)  // 8192
#define LT2  (BB*SA2N)  // 2048

// TDM-deposited tile row stride in halfs: 8 DWORDs data + 2 DWORDs pad = 10 DWORDs
#define TDM_ROWH 20

// ---------------- per-batch mean of xyz ----------------
__global__ __launch_bounds__(256)
void batch_mean_kernel(const float* __restrict__ xyz, float* __restrict__ meanb) {
    __shared__ float sx[256], sy[256], sz[256];
    int b = blockIdx.x;
    const float* p = xyz + (size_t)b * NN * 3;
    float ax = 0.f, ay = 0.f, az = 0.f;
    for (int i = threadIdx.x; i < NN; i += 256) {
        ax += p[i*3+0]; ay += p[i*3+1]; az += p[i*3+2];
    }
    sx[threadIdx.x] = ax; sy[threadIdx.x] = ay; sz[threadIdx.x] = az;
    __syncthreads();
    for (int s = 128; s > 0; s >>= 1) {
        if (threadIdx.x < s) {
            sx[threadIdx.x] += sx[threadIdx.x+s];
            sy[threadIdx.x] += sy[threadIdx.x+s];
            sz[threadIdx.x] += sz[threadIdx.x+s];
        }
        __syncthreads();
    }
    if (threadIdx.x == 0) {
        meanb[b*3+0] = sx[0] / (float)NN;
        meanb[b*3+1] = sy[0] / (float)NN;
        meanb[b*3+2] = sz[0] / (float)NN;
    }
}

// ---------------- positional encoding: pe[9][B*N] (f32 master copy) ----------------
__global__ __launch_bounds__(256)
void pos_enc_kernel(const float* __restrict__ xyz, const float* __restrict__ meanb,
                    float* __restrict__ pe) {
    int t = blockIdx.x * 256 + threadIdx.x;
    if (t >= LT0) return;
    int b = t >> 13;
    float x = xyz[(size_t)t*3+0], y = xyz[(size_t)t*3+1], z = xyz[(size_t)t*3+2];
    float cx = x - meanb[b*3+0], cy = y - meanb[b*3+1], cz = z - meanb[b*3+2];
    float inv = 1.f / (sqrtf(cx*cx + cy*cy + cz*cz) + 1e-8f);
    pe[(size_t)0*LT0 + t] = x;
    pe[(size_t)1*LT0 + t] = y;
    pe[(size_t)2*LT0 + t] = z;
    pe[(size_t)3*LT0 + t] = cx;
    pe[(size_t)4*LT0 + t] = cy;
    pe[(size_t)5*LT0 + t] = cz;
    pe[(size_t)6*LT0 + t] = cx * inv;
    pe[(size_t)7*LT0 + t] = cy * inv;
    pe[(size_t)8*LT0 + t] = cz * inv;
}

// ---------------- farthest point sampling (one block per batch) ----------------
__global__ __launch_bounds__(256)
void fps_kernel(const float* __restrict__ pts, int n, int npoint, int* __restrict__ idx_out) {
    __shared__ float dist[8192];
    __shared__ float rmax[256];
    __shared__ int   rarg[256];
    __shared__ int   s_far;
    int b = blockIdx.x;
    const float* p = pts + (size_t)b * n * 3;
    for (int i = threadIdx.x; i < n; i += 256) dist[i] = 1e10f;
    int far = 0;
    __syncthreads();
    for (int s = 0; s < npoint; ++s) {
        if (threadIdx.x == 0) idx_out[b * npoint + s] = far;
        float fx = p[far*3+0], fy = p[far*3+1], fz = p[far*3+2];
        float best = -1.f; int barg = 0;
        for (int i = threadIdx.x; i < n; i += 256) {
            float dx = p[i*3+0]-fx, dy = p[i*3+1]-fy, dz = p[i*3+2]-fz;
            float d = dx*dx + dy*dy + dz*dz;
            float nd = fminf(dist[i], d);
            dist[i] = nd;
            if (nd > best) { best = nd; barg = i; }
        }
        rmax[threadIdx.x] = best; rarg[threadIdx.x] = barg;
        __syncthreads();
        for (int st = 128; st > 0; st >>= 1) {
            if (threadIdx.x < st && rmax[threadIdx.x+st] > rmax[threadIdx.x]) {
                rmax[threadIdx.x] = rmax[threadIdx.x+st];
                rarg[threadIdx.x] = rarg[threadIdx.x+st];
            }
            __syncthreads();
        }
        if (threadIdx.x == 0) s_far = rarg[0];
        __syncthreads();
        far = s_far;
    }
}

// ---------------- pack f32 weights -> padded f16 [Cout][CinP] ----------------
__global__ __launch_bounds__(256)
void pack_w_kernel(const float* __restrict__ W, _Float16* __restrict__ Wh,
                   int Cin, int CinP, int total) {
    int t = blockIdx.x * 256 + threadIdx.x;
    if (t >= total) return;
    int o = t / CinP, k = t - o * CinP;
    Wh[t] = (k < Cin) ? (_Float16)W[(size_t)o * Cin + k] : (_Float16)0.f;
}

// ---------------- gather l1: xyz subset + padded f16 features (32 rows) ----------------
__global__ __launch_bounds__(256)
void gather_l1_kernel(const float* __restrict__ xyz, const float* __restrict__ pe,
                      const int* __restrict__ idx1,
                      float* __restrict__ l1xyz, _Float16* __restrict__ l1pts_h) {
    int t = blockIdx.x * 256 + threadIdx.x;
    if (t >= LT1) return;
    int b = t >> 9;
    int src = idx1[t];
    size_t col = (size_t)b * NN + src;
    l1xyz[(size_t)t*3+0] = xyz[col*3+0];
    l1xyz[(size_t)t*3+1] = xyz[col*3+1];
    l1xyz[(size_t)t*3+2] = xyz[col*3+2];
    #pragma unroll
    for (int c = 0; c < 32; ++c) {
        _Float16 v = (c < 9) ? (_Float16)pe[(size_t)c*LT0 + col] : (_Float16)0.f;
        l1pts_h[(size_t)c*LT1 + t] = v;
    }
}

// ---------------- gather l2 points (f16) from l1_feat via idx2 ----------------
__global__ __launch_bounds__(256)
void gather_l2_kernel(const _Float16* __restrict__ l1feat_h, const int* __restrict__ idx2,
                      _Float16* __restrict__ l2pts_h) {
    int t = blockIdx.x * 256 + threadIdx.x;
    if (t >= 128 * LT2) return;
    int c   = t >> 11;
    int col = t & (LT2 - 1);
    int b   = col >> 7;
    int j   = col & 127;
    int src = b * SA1N + idx2[b * SA2N + j];
    l2pts_h[(size_t)c*LT2 + col] = l1feat_h[(size_t)c*LT1 + src];
}

#if USE_TDM
// Issue one TDM descriptor: load a 32(K) x 16(N) f16 tile (row-major in memory,
// row stride Ltot halfs) into LDS at ldsOff, padding each 8-DWORD row with
// 2 DWORDs (pad_interval=2, pad_amount=1) -> LDS row stride 10 DWORDs, which
// makes the per-lane strided B-operand reads bank-conflict free.
__device__ __forceinline__ void tdm_issue_tile(const _Float16* gsrc, unsigned ldsOff,
                                               int CinP, int Ltot) {
    unsigned long long ga = (unsigned long long)(const void*)gsrc;
    // D# group 0: count=1 | lds_addr | global_addr | type=2
    v4u g0 = { 1u, ldsOff, (unsigned)(ga & 0xFFFFFFFFu),
               (unsigned)((ga >> 32) & 0x01FFFFFFu) | (2u << 30) };
    // D# group 1: data_size=2B | pad_enable | pad_interval=8DW | pad_amount=2DW;
    //             tensor_dim0=Ltot; tensor_dim1=CinP; tile_dim0=16; tile_dim1=32;
    //             tensor_dim0_stride=Ltot
    v8i g1 = { (int)0x02910000u,
               (int)(((unsigned)Ltot & 0xFFFFu) << 16),
               (int)(((unsigned)Ltot >> 16) | (((unsigned)CinP & 0xFFFFu) << 16)),
               (int)(16u << 16),
               (int)32,
               (int)Ltot,
               0, 0 };
    v4i gz = { 0, 0, 0, 0 };
#if TDM_SIX_ARG
    v8i gz8 = { 0, 0, 0, 0, 0, 0, 0, 0 };
    __builtin_amdgcn_tensor_load_to_lds(g0, g1, gz, gz, gz8, 0);
#else
    __builtin_amdgcn_tensor_load_to_lds(g0, g1, gz, gz, 0);
#endif
}
#endif

// ---------------- WMMA GEMM: Y[Cout,Lt] = Wh[Cout,CinP] * Xh[CinP,Lt] + bias ----------------
// One block covers full Cout (wave w -> rows 16w..16w+15) and 16 columns, so X is
// streamed from HBM exactly once. The 32(K)x16(N) B tiles are double-buffered in
// LDS by the Tensor Data Mover: wave 0 waits for tile i (s_wait_tensorcnt 0),
// a workgroup barrier publishes it, then wave 0 immediately issues the DMA for
// tile i+1 into the other buffer while all waves compute tile i.
// A operand = two aligned global_load_b128 from packed f16 weights.
__global__ __launch_bounds__(512)
void gemm_f16_wmma_kernel(const _Float16* __restrict__ Wh, const float* __restrict__ bias,
                          const _Float16* __restrict__ Xh, float* __restrict__ Y,
                          int CinP, int Ltot) {
    const int tid  = threadIdx.x;
    const int lane = tid & 31, wave = tid >> 5;
    const int laneHi = lane >> 4, lanelo = lane & 15;
    const int m0 = wave * 16;
    const int n0 = blockIdx.x * 16;
    const int Mrow = m0 + lanelo;
    const int Ncol = n0 + lanelo;
    v8f acc = {};
#if USE_TDM
    __shared__ __attribute__((aligned(16))) _Float16 Bs[2][32 * TDM_ROWH];
    const int nIter = CinP >> 5;
    if (wave == 0)
        tdm_issue_tile(Xh + n0, (unsigned)(unsigned long long)(const void*)&Bs[0][0],
                       CinP, Ltot);
    for (int it = 0; it < nIter; ++it) {
        if (wave == 0) __builtin_amdgcn_s_wait_tensorcnt(0);  // tile `it` landed
        __syncthreads();                                      // publish to all waves
        if (wave == 0 && it + 1 < nIter)                      // overlap DMA of next tile
            tdm_issue_tile(Xh + (size_t)(it + 1) * 32 * Ltot + n0,
                           (unsigned)(unsigned long long)(const void*)&Bs[(it + 1) & 1][0],
                           CinP, Ltot);
        const _Float16* bs = &Bs[it & 1][0];
        // A (16x32 f16): element e -> K = 16*laneHi + e
        const v8h* ar = (const v8h*)(Wh + (size_t)Mrow * CinP + ((size_t)it << 5));
        v8h a0 = ar[2 * laneHi];
        v8h a1 = ar[2 * laneHi + 1];
        v16h av = __builtin_shufflevector(a0, a1, 0,1,2,3,4,5,6,7,8,9,10,11,12,13,14,15);
        // B (32x16 f16): element e -> K = 16*laneHi + e, N = lanelo
        v16h bv;
        #pragma unroll
        for (int e = 0; e < 16; ++e)
            bv[e] = bs[(16 * laneHi + e) * TDM_ROWH + lanelo];  // imm-offset ds_load_u16
        acc = __builtin_amdgcn_wmma_f32_16x16x32_f16(
                  false, av, false, bv, (short)0, acc, false, false);
    }
#else
    __shared__ __attribute__((aligned(16))) _Float16 Bs[32 * 24];   // B-operand swizzled
    const int T = blockDim.x;
    for (int k0 = 0; k0 < CinP; k0 += 32) {
        __syncthreads();
        // cooperative staging, pre-swizzled to B-operand layout:
        // Bs[((k>>4)*16 + n)*24 + (k&15)] = X[k0+k][n0+n]
        for (int idx = tid; idx < 512; idx += T) {
            int k = idx >> 4, n = idx & 15;
            int ln = ((k >> 4) << 4) + n;
            Bs[ln * 24 + (k & 15)] = Xh[(size_t)(k0 + k) * Ltot + n0 + n];
        }
        __syncthreads();
        if (k0 + 32 < CinP)
            __builtin_prefetch(Xh + (size_t)(k0 + 32 + (tid >> 4)) * Ltot + n0 + (tid & 15), 0, 3);
        const v8h* ar = (const v8h*)(Wh + (size_t)Mrow * CinP + k0);
        v8h a0 = ar[2 * laneHi];
        v8h a1 = ar[2 * laneHi + 1];
        v16h av = __builtin_shufflevector(a0, a1, 0,1,2,3,4,5,6,7,8,9,10,11,12,13,14,15);
        const v8h* br = (const v8h*)(Bs + lane * 24);
        v8h b0 = br[0];
        v8h b1 = br[1];
        v16h bv = __builtin_shufflevector(b0, b1, 0,1,2,3,4,5,6,7,8,9,10,11,12,13,14,15);
        acc = __builtin_amdgcn_wmma_f32_16x16x32_f16(
                  false, av, false, bv, (short)0, acc, false, false);
    }
#endif
    // D (16x16 f32): element r -> M = r + 8*laneHi, N = lanelo
    #pragma unroll
    for (int r = 0; r < 8; ++r) {
        int M = m0 + r + 8 * laneHi;
        Y[(size_t)M * Ltot + Ncol] = acc[r] + bias[M];
    }
}

// ---------------- BN stats: mean/var per channel over Ltot columns ----------------
__global__ __launch_bounds__(256)
void bn_stats_kernel(const float* __restrict__ Y, float* __restrict__ stats,
                     int Cout, int Ltot) {
    __shared__ float s1[256], s2[256];
    int c = blockIdx.x;
    float a = 0.f, b = 0.f;
    for (int i = threadIdx.x; i < Ltot; i += 256) {
        float v = Y[(size_t)c * Ltot + i];
        a += v; b += v * v;
    }
    s1[threadIdx.x] = a; s2[threadIdx.x] = b;
    __syncthreads();
    for (int s = 128; s > 0; s >>= 1) {
        if (threadIdx.x < s) {
            s1[threadIdx.x] += s1[threadIdx.x+s];
            s2[threadIdx.x] += s2[threadIdx.x+s];
        }
        __syncthreads();
    }
    if (threadIdx.x == 0) {
        float mean = s1[0] / (float)Ltot;
        float var  = s2[0] / (float)Ltot - mean * mean;
        stats[c]        = mean;
        stats[Cout + c] = var;
    }
}

// ---------------- BN apply + ReLU -> f16 next-layer input ----------------
__global__ __launch_bounds__(256)
void bn_apply_relu_f16_kernel(const float* __restrict__ Y, _Float16* __restrict__ Xout,
                              const float* __restrict__ stats,
                              const float* __restrict__ gamma, const float* __restrict__ beta,
                              int Cout, int Ltot) {
    size_t t = (size_t)blockIdx.x * 256 + threadIdx.x;
    if (t >= (size_t)Cout * Ltot) return;
    int c = (int)(t / (size_t)Ltot);
    float v = (Y[t] - stats[c]) * rsqrtf(stats[Cout + c] + 1e-5f) * gamma[c] + beta[c];
    Xout[t] = (_Float16)(v > 0.f ? v : 0.f);
}

// ---------------- BN apply + ReLU in place (final f32 layer) ----------------
__global__ __launch_bounds__(256)
void bn_apply_relu_f32_kernel(float* __restrict__ Y, const float* __restrict__ stats,
                              const float* __restrict__ gamma, const float* __restrict__ beta,
                              int Cout, int Ltot) {
    size_t t = (size_t)blockIdx.x * 256 + threadIdx.x;
    if (t >= (size_t)Cout * Ltot) return;
    int c = (int)(t / (size_t)Ltot);
    float v = (Y[t] - stats[c]) * rsqrtf(stats[Cout + c] + 1e-5f) * gamma[c] + beta[c];
    Y[t] = v > 0.f ? v : 0.f;
}

// ---------------- concat fp2 (f16): [l1_feat(128); upsample(l2_feat)(128)] ----------------
__global__ __launch_bounds__(256)
void concat_fp2_kernel(const _Float16* __restrict__ l1feat_h, const _Float16* __restrict__ l2feat_h,
                       _Float16* __restrict__ out) {
    int t = blockIdx.x * 256 + threadIdx.x;
    if (t >= 256 * LT1) return;
    int r   = t >> 13;
    int col = t & (LT1 - 1);
    int b   = col >> 9;
    int j   = col & 511;
    _Float16 v;
    if (r < 128) v = l1feat_h[(size_t)r * LT1 + col];
    else         v = l2feat_h[(size_t)(r - 128) * LT2 + b * SA2N + (j >> 2)];
    out[(size_t)r * LT1 + col] = v;
}

// ---------------- concat fp1 (f16, 288 rows): [upsample(fp2_out)(256); l0=pe(9); pad(23)] ----------------
__global__ __launch_bounds__(256)
void concat_fp1_kernel(const _Float16* __restrict__ fp2out_h, const float* __restrict__ pe,
                       _Float16* __restrict__ out) {
    size_t t = (size_t)blockIdx.x * 256 + threadIdx.x;
    if (t >= (size_t)288 * LT0) return;
    int r   = (int)(t / (size_t)LT0);
    int col = (int)(t % (size_t)LT0);
    int b   = col >> 13;
    int l   = col & (NN - 1);
    _Float16 v;
    if (r < 256)      v = fp2out_h[(size_t)r * LT1 + b * SA1N + (l >> 4)];
    else if (r < 265) v = (_Float16)pe[(size_t)(r - 256) * LT0 + col];
    else              v = (_Float16)0.f;
    out[(size_t)r * LT0 + col] = v;
}

// ---------------- final transpose: out[b][l][c] = Y[c][b*N+l] ----------------
__global__ __launch_bounds__(256)
void transpose_out_kernel(const float* __restrict__ Y, float* __restrict__ out) {
    size_t t = (size_t)blockIdx.x * 256 + threadIdx.x;
    if (t >= (size_t)BB * NN * CEMB) return;
    int c = (int)(t & (CEMB - 1));
    int l = (int)((t >> 7) & (NN - 1));
    int b = (int)(t >> 20);
    out[t] = Y[(size_t)c * LT0 + (size_t)b * NN + l];
}

// ---------------- host-side layer driver ----------------
static void run_layer(const _Float16* Wh, const float* bias, const float* g, const float* be,
                      const _Float16* Xh, float* Yf32, _Float16* XoutH, float* stats,
                      int Cout, int CinP, int Ltot, bool lastF32, hipStream_t s) {
    dim3 grid(Ltot / 16);
    int T = (Cout / 16) * 32;                 // one wave per 16 output rows
    gemm_f16_wmma_kernel<<<grid, T, 0, s>>>(Wh, bias, Xh, Yf32, CinP, Ltot);
    bn_stats_kernel<<<Cout, 256, 0, s>>>(Yf32, stats, Cout, Ltot);
    size_t tot = (size_t)Cout * Ltot;
    if (lastF32)
        bn_apply_relu_f32_kernel<<<(unsigned)((tot + 255) / 256), 256, 0, s>>>(Yf32, stats, g, be, Cout, Ltot);
    else
        bn_apply_relu_f16_kernel<<<(unsigned)((tot + 255) / 256), 256, 0, s>>>(Yf32, XoutH, stats, g, be, Cout, Ltot);
}

extern "C" void kernel_launch(void* const* d_in, const int* in_sizes, int n_in,
                              void* d_out, int out_size, void* d_ws, size_t ws_size,
                              hipStream_t stream) {
    const float* xyz = (const float*)d_in[0];
    auto F = [&](int i) { return (const float*)d_in[i]; };
    // sa1: d_in[1..12], sa2: d_in[13..24], fp2: d_in[25..32], fp1: d_in[33..40]

    float* ws = (float*)d_ws;
    size_t o = 0;
    float* pe     = ws + o;  o += (size_t)9 * LT0;        // f32 master pos-enc
    float* meanb  = ws + o;  o += 64;
    float* l1xyz  = ws + o;  o += (size_t)BB * SA1N * 3;
    int*   idx1   = (int*)(ws + o);  o += BB * SA1N;
    int*   idx2   = (int*)(ws + o);  o += BB * SA2N;
    float* stats  = ws + o;  o += 512;
    float* Yf32   = ws + o;  o += (size_t)128 * LT0;      // shared f32 GEMM output (max size)
    _Float16* WhA = (_Float16*)(ws + o);  o += 131072;    // packed weight arena (<=247808 halfs)
    _Float16* l1pts_h  = (_Float16*)(ws + o); o += (size_t)32  * LT1 / 2;
    _Float16* sa1a_h   = (_Float16*)(ws + o); o += (size_t)64  * LT1 / 2;
    _Float16* sa1b_h   = (_Float16*)(ws + o); o += (size_t)64  * LT1 / 2;
    _Float16* l1feat_h = (_Float16*)(ws + o); o += (size_t)128 * LT1 / 2;
    _Float16* l2pts_h  = (_Float16*)(ws + o); o += (size_t)128 * LT2 / 2;
    _Float16* sa2a_h   = (_Float16*)(ws + o); o += (size_t)128 * LT2 / 2;
    _Float16* sa2b_h   = (_Float16*)(ws + o); o += (size_t)128 * LT2 / 2;
    _Float16* l2feat_h = (_Float16*)(ws + o); o += (size_t)128 * LT2 / 2;
    _Float16* fp2in_h  = (_Float16*)(ws + o); o += (size_t)256 * LT1 / 2;
    _Float16* fp2a_h   = (_Float16*)(ws + o); o += (size_t)256 * LT1 / 2;
    _Float16* fp2out_h = (_Float16*)(ws + o); o += (size_t)256 * LT1 / 2;
    _Float16* fp1in_h  = (_Float16*)(ws + o); o += (size_t)288 * LT0 / 2;
    _Float16* fp1a_h   = (_Float16*)(ws + o); o += (size_t)128 * LT0 / 2;

    // ---- pack all weights to padded f16 ----
    size_t wo = 0;
    auto packW = [&](int i, int Cout, int Cin, int CinP) -> const _Float16* {
        _Float16* dst = WhA + wo;
        int tot = Cout * CinP;
        pack_w_kernel<<<(tot + 255) / 256, 256, 0, stream>>>(F(i), dst, Cin, CinP, tot);
        wo += (size_t)tot;
        return dst;
    };
    const _Float16* w_sa1_0 = packW(1,   64,   9,  32);
    const _Float16* w_sa1_1 = packW(5,   64,  64,  64);
    const _Float16* w_sa1_2 = packW(9,  128,  64,  64);
    const _Float16* w_sa2_0 = packW(13, 128, 128, 128);
    const _Float16* w_sa2_1 = packW(17, 128, 128, 128);
    const _Float16* w_sa2_2 = packW(21, 128, 128, 128);
    const _Float16* w_fp2_0 = packW(25, 256, 256, 256);
    const _Float16* w_fp2_1 = packW(29, 256, 256, 256);
    const _Float16* w_fp1_0 = packW(33, 128, 265, 288);
    const _Float16* w_fp1_1 = packW(37, 128, 128, 128);

    // 1) per-batch mean + positional encoding
    batch_mean_kernel<<<BB, 256, 0, stream>>>(xyz, meanb);
    pos_enc_kernel<<<(LT0 + 255) / 256, 256, 0, stream>>>(xyz, meanb, pe);

    // 2) FPS level 1 (8192 -> 512) + gather (pads 9 -> 32 channels)
    fps_kernel<<<BB, 256, 0, stream>>>(xyz, NN, SA1N, idx1);
    gather_l1_kernel<<<(LT1 + 255) / 256, 256, 0, stream>>>(xyz, pe, idx1, l1xyz, l1pts_h);

    // 3) SA1 MLP: 9(32) -> 64 -> 64 -> 128 over Ltot=8192
    run_layer(w_sa1_0, F(2),  F(3),  F(4),  l1pts_h, Yf32, sa1a_h,   stats,  64,  32, LT1, false, stream);
    run_layer(w_sa1_1, F(6),  F(7),  F(8),  sa1a_h,  Yf32, sa1b_h,   stats,  64,  64, LT1, false, stream);
    run_layer(w_sa1_2, F(10), F(11), F(12), sa1b_h,  Yf32, l1feat_h, stats, 128,  64, LT1, false, stream);

    // 4) FPS level 2 (512 -> 128) + gather
    fps_kernel<<<BB, 256, 0, stream>>>(l1xyz, SA1N, SA2N, idx2);
    gather_l2_kernel<<<(128 * LT2 + 255) / 256, 256, 0, stream>>>(l1feat_h, idx2, l2pts_h);

    // 5) SA2 MLP: 128 -> 128 -> 128 -> 128 over Ltot=2048
    run_layer(w_sa2_0, F(14), F(15), F(16), l2pts_h, Yf32, sa2a_h,   stats, 128, 128, LT2, false, stream);
    run_layer(w_sa2_1, F(18), F(19), F(20), sa2a_h,  Yf32, sa2b_h,   stats, 128, 128, LT2, false, stream);
    run_layer(w_sa2_2, F(22), F(23), F(24), sa2b_h,  Yf32, l2feat_h, stats, 128, 128, LT2, false, stream);

    // 6) FP2: concat [l1_feat; upsample(l2_feat)] then 256 -> 256 -> 256
    concat_fp2_kernel<<<(256 * LT1 + 255) / 256, 256, 0, stream>>>(l1feat_h, l2feat_h, fp2in_h);
    run_layer(w_fp2_0, F(26), F(27), F(28), fp2in_h, Yf32, fp2a_h,   stats, 256, 256, LT1, false, stream);
    run_layer(w_fp2_1, F(30), F(31), F(32), fp2a_h,  Yf32, fp2out_h, stats, 256, 256, LT1, false, stream);

    // 7) FP1: concat [upsample(fp2_out); l0; pad] then 265(288) -> 128 -> 128 over Ltot=131072
    {
        size_t tot = (size_t)288 * LT0;
        concat_fp1_kernel<<<(unsigned)((tot + 255) / 256), 256, 0, stream>>>(fp2out_h, pe, fp1in_h);
    }
    run_layer(w_fp1_0, F(34), F(35), F(36), fp1in_h, Yf32, fp1a_h, stats, 128, 288, LT0, false, stream);
    run_layer(w_fp1_1, F(38), F(39), F(40), fp1a_h,  Yf32, nullptr, stats, 128, 128, LT0, true,  stream);

    // 8) transpose to (B, N, 128)
    {
        size_t tot = (size_t)BB * NN * CEMB;
        transpose_out_kernel<<<(unsigned)((tot + 255) / 256), 256, 0, stream>>>(Yf32, (float*)d_out);
    }
    (void)in_sizes; (void)n_in; (void)out_size; (void)ws_size;
}
